// MixProp_15066745275102
// MI455X (gfx1250) — matrix operand order repaired and checked by reading, not compiled, not run on hardware
//
#include <hip/hip_runtime.h>

typedef __bf16  v16bf __attribute__((ext_vector_type(16)));
typedef float   v8f   __attribute__((ext_vector_type(8)));
typedef unsigned int   u32;
typedef unsigned short u16;

#define NNODES 512
#define SEQ    256
#define CIN    32
#define BATCH  8
#define BETA   0.05f

static __device__ __forceinline__ u16 f2bf(float f) {
    u32 u = __float_as_uint(f);
    u += 0x7FFFu + ((u >> 16) & 1u);   // round-to-nearest-even
    return (u16)(u >> 16);
}
static __device__ __forceinline__ u32 pk2(float lo, float hi) {
    return (u32)f2bf(lo) | ((u32)f2bf(hi) << 16);
}
static __device__ __forceinline__ float bf2f(u16 v) {
    return __uint_as_float(((u32)v) << 16);
}
// pack two already-bf16 values (pure integer merge, ~2 VALU)
#define PKU(a, b) ((u32)(a) | ((u32)(b) << 16))

// ---------------------------------------------------------------------------
// Kernel 1: A_norm[v][w] = (A[v][w] + (v==w)) / rowsum   (fp32, ws)
// ---------------------------------------------------------------------------
__global__ __launch_bounds__(256) void mp_normalize(const float* __restrict__ A,
                                                    float* __restrict__ An) {
    __shared__ float red[256];
    const int row = blockIdx.x;
    const int t = threadIdx.x;
    float v0 = A[row * NNODES + t]       + (t == row ? 1.0f : 0.0f);
    float v1 = A[row * NNODES + t + 256] + ((t + 256) == row ? 1.0f : 0.0f);
    red[t] = v0 + v1;
    __syncthreads();
    for (int s = 128; s > 0; s >>= 1) {
        if (t < s) red[t] += red[t + s];
        __syncthreads();
    }
    const float inv = 1.0f / red[0];
    An[row * NNODES + t]       = v0 * inv;
    An[row * NNODES + t + 256] = v1 * inv;
}

// ---------------------------------------------------------------------------
// Kernel 1b: pre-swizzle A_norm into wave32 A-fragment order (bf16).
// slot j -> K = ((j&8)<<1) + half*8 + (j&7);  grid = (16 kblocks, 32 vtiles)
// ---------------------------------------------------------------------------
__global__ __launch_bounds__(32) void mp_prep_afrag(const float* __restrict__ An,
                                                    u16* __restrict__ Afrag) {
    const int kblock = blockIdx.x, vtile = blockIdx.y;
    const int lane = threadIdx.x, half = lane >> 4, lidx = lane & 15;
    const float* ar = An + (size_t)(vtile * 16 + lidx) * NNODES + kblock * 32;
    const int k0 = half * 8, k1 = 16 + half * 8;
    uint4 q0 = make_uint4(pk2(ar[k0+0], ar[k0+1]), pk2(ar[k0+2], ar[k0+3]),
                          pk2(ar[k0+4], ar[k0+5]), pk2(ar[k0+6], ar[k0+7]));
    uint4 q1 = make_uint4(pk2(ar[k1+0], ar[k1+1]), pk2(ar[k1+2], ar[k1+3]),
                          pk2(ar[k1+4], ar[k1+5]), pk2(ar[k1+6], ar[k1+7]));
    u16* dst = Afrag + (((size_t)vtile * 16 + kblock) << 9) + (lane << 4);
    *(uint4*)dst       = q0;
    *(uint4*)(dst + 8) = q1;
}

// ---------------------------------------------------------------------------
// Kernel 1c: pre-swizzle W (64x128 fp32) into A-fragment order (bf16). grid=(4,4)
// ---------------------------------------------------------------------------
__global__ __launch_bounds__(32) void mp_prep_wfrag(const float* __restrict__ W,
                                                    u16* __restrict__ Wfrag) {
    const int kblock = blockIdx.x, mtile = blockIdx.y;
    const int lane = threadIdx.x, half = lane >> 4, lidx = lane & 15;
    const float* wr = W + (size_t)(mtile * 16 + lidx) * 128 + kblock * 32;
    const int k0 = half * 16, k1 = half * 16 + 8;
    uint4 q0 = make_uint4(pk2(wr[k0+0], wr[k0+1]), pk2(wr[k0+2], wr[k0+3]),
                          pk2(wr[k0+4], wr[k0+5]), pk2(wr[k0+6], wr[k0+7]));
    uint4 q1 = make_uint4(pk2(wr[k1+0], wr[k1+1]), pk2(wr[k1+2], wr[k1+3]),
                          pk2(wr[k1+4], wr[k1+5]), pk2(wr[k1+6], wr[k1+7]));
    u16* dst = Wfrag + (((size_t)mtile * 4 + kblock) << 9) + (lane << 4);
    *(uint4*)dst       = q0;
    *(uint4*)(dst + 8) = q1;
}

// ---------------------------------------------------------------------------
// Kernel 1d: X (fp32) -> Xb (bf16), same [b,c,w,l] layout. 8 elems/thread.
// ---------------------------------------------------------------------------
__global__ __launch_bounds__(256) void mp_xcopy(const float* __restrict__ X,
                                                u16* __restrict__ Xb) {
    const size_t i = ((size_t)blockIdx.x * 256 + threadIdx.x) * 8;
    float4 a = *(const float4*)(X + i);
    float4 b = *(const float4*)(X + i + 4);
    uint4 q = make_uint4(pk2(a.x, a.y), pk2(a.z, a.w), pk2(b.x, b.y), pk2(b.z, b.w));
    *(uint4*)(Xb + i) = q;
}

// ---------------------------------------------------------------------------
// Kernel 2: one propagation hop (all tensors bf16).
//   Hout = bf16( BETA*Xb + (1-BETA) * sum_w An[v,w]*Hprev[b,c,w,l] )
// 8 waves, macro-tile 128(v) x 64(l); A frags from global (L2-hot),
// B panel double-buffered in LDS (pure u16 repack, no float cvt). grid=(16,256)
// ---------------------------------------------------------------------------
__global__ __launch_bounds__(256) void mixprop_hop(const u16* __restrict__ Afrag,
                                                   const u16* __restrict__ Hprev,
                                                   const u16* __restrict__ Xb,
                                                   u16* __restrict__ Hout) {
    __shared__ __align__(16) u16 ldsB[2][4 * 512]; // 2 x (4 subtiles of 32x16 bf16)

    const int t    = threadIdx.x;
    const int lane = t & 31;
    const int wv   = t >> 5;           // wave id = v-tile within macro-tile
    const int half = lane >> 4;
    const int lidx = lane & 15;

    const size_t sbase = (size_t)blockIdx.y * (NNODES * SEQ);
    const int vbase = (blockIdx.x & 3) * 128;
    const int lbase = (blockIdx.x >> 2) * 64;
    const int vtile = (vbase >> 4) + wv;

    // B-staging personality (constant per thread)
    const int sj   = t >> 7;             // which 8 j-slots this thread fills
    const int ss   = (t >> 5) & 3;       // l-subtile
    const int scol = lbase + ss * 16 + lidx;
    const size_t srow0 = (size_t)(half * 8 + sj * 16) * SEQ;

    v8f acc[4];
#pragma unroll
    for (int i = 0; i < 4; ++i) acc[i] = (v8f){0.f,0.f,0.f,0.f,0.f,0.f,0.f,0.f};

#define STAGE_B(KK, BUF)                                                           \
    {                                                                              \
        const u16* hc = Hprev + sbase + (size_t)(KK) * 32 * SEQ + srow0 + scol;    \
        uint4 q = make_uint4(PKU(hc[0*SEQ], hc[1*SEQ]), PKU(hc[2*SEQ], hc[3*SEQ]), \
                             PKU(hc[4*SEQ], hc[5*SEQ]), PKU(hc[6*SEQ], hc[7*SEQ]));\
        *(uint4*)&ldsB[BUF][(ss << 9) + (lane << 4) + (sj << 3)] = q;              \
    }

    STAGE_B(0, 0);
    for (int kk = 0; kk < 16; ++kk) {
        __syncthreads();
        if (kk < 15) STAGE_B(kk + 1, (kk + 1) & 1);
        const v16bf aF = *(const v16bf*)(Afrag + (((size_t)vtile * 16 + kk) << 9) + (lane << 4));
        const u16* bb = &ldsB[kk & 1][lane << 4];
#pragma unroll
        for (int lt = 0; lt < 4; ++lt) {
            const v16bf bF = *(const v16bf*)(bb + (lt << 9));
            acc[lt] = __builtin_amdgcn_wmma_f32_16x16x32_bf16(
                false, aF, false, bF, (short)0, acc[lt], false, false);
        }
    }
#undef STAGE_B

    // epilogue: blend + convert once per output (C/D layout: VGPR r -> M=half*8+r, N=lidx)
#pragma unroll
    for (int lt = 0; lt < 4; ++lt) {
        const int col = lbase + lt * 16 + lidx;
#pragma unroll
        for (int r = 0; r < 8; ++r) {
            const int row = vbase + wv * 16 + half * 8 + r;
            const size_t idx = sbase + (size_t)row * SEQ + col;
            const float f = BETA * bf2f(Xb[idx]) + (1.0f - BETA) * acc[lt][r];
            Hout[idx] = f2bf(f);
        }
    }
}

// ---------------------------------------------------------------------------
// Kernel 3: out[b,o,p] = bias[o] + sum_{c<128} W[o,c] * H0[b,c,p]
// Channel blocks: 0 -> Xb, 1..3 -> hop buffers (all bf16). grid = (1024, 8)
// ---------------------------------------------------------------------------
__global__ __launch_bounds__(256) void mixprop_conv(const u16* __restrict__ Xb,
                                                    const u16* __restrict__ h1,
                                                    const u16* __restrict__ h2,
                                                    const u16* __restrict__ h3,
                                                    const u16* __restrict__ Wfrag,
                                                    const float* __restrict__ bias,
                                                    float* __restrict__ out) {
    __shared__ __align__(16) u16 ldsH[2][8 * 512]; // 2 x (8 subtiles of 32x16 bf16)

    const int t    = threadIdx.x;
    const int lane = t & 31;
    const int wv   = t >> 5;
    const int half = lane >> 4;
    const int lidx = lane & 15;

    const int b     = blockIdx.y;
    const int pbase = blockIdx.x * 128;
    const size_t NL = (size_t)NNODES * SEQ;

    const u16* srcs[4] = {Xb, h1, h2, h3};

    v8f acc[4];
#pragma unroll
    for (int i = 0; i < 4; ++i) acc[i] = (v8f){0.f,0.f,0.f,0.f,0.f,0.f,0.f,0.f};

    const int ss = t >> 5;                       // p-subtile for staging
    const int sp = pbase + ss * 16 + lidx;
    const size_t schan = ((size_t)b * CIN + half * 16) * NL;

#define STAGE_H(KK, BUF)                                                             \
    {                                                                                \
        const u16* hp = srcs[KK] + schan + sp;                                       \
        uint4 q0 = make_uint4(PKU(hp[0*NL],  hp[1*NL]),  PKU(hp[2*NL],  hp[3*NL]),   \
                              PKU(hp[4*NL],  hp[5*NL]),  PKU(hp[6*NL],  hp[7*NL]));  \
        uint4 q1 = make_uint4(PKU(hp[8*NL],  hp[9*NL]),  PKU(hp[10*NL], hp[11*NL]),  \
                              PKU(hp[12*NL], hp[13*NL]), PKU(hp[14*NL], hp[15*NL])); \
        u16* dst = &ldsH[BUF][(ss << 9) + (lane << 4)];                              \
        *(uint4*)dst       = q0;                                                     \
        *(uint4*)(dst + 8) = q1;                                                     \
    }

    const int mt = wv & 3;
    const int pg = wv >> 2;

    STAGE_H(0, 0);
#pragma unroll
    for (int kk = 0; kk < 4; ++kk) {
        __syncthreads();
        if (kk < 3) STAGE_H(kk + 1, (kk + 1) & 1);
        const v16bf aF = *(const v16bf*)(Wfrag + (((size_t)mt * 4 + kk) << 9) + (lane << 4));
        const u16* bb = &ldsH[kk & 1][lane << 4];
#pragma unroll
        for (int lt = 0; lt < 4; ++lt) {
            const v16bf bF = *(const v16bf*)(bb + ((pg * 4 + lt) << 9));
            acc[lt] = __builtin_amdgcn_wmma_f32_16x16x32_bf16(
                false, aF, false, bF, (short)0, acc[lt], false, false);
        }
    }
#undef STAGE_H

    // epilogue with fused bias (fp32 output)
#pragma unroll
    for (int lt = 0; lt < 4; ++lt) {
        const int p = pbase + (pg * 4 + lt) * 16 + lidx;
#pragma unroll
        for (int r = 0; r < 8; ++r) {
            const int o = mt * 16 + half * 8 + r;
            out[((size_t)b * 64 + o) * NL + p] = acc[lt][r] + bias[o];
        }
    }
}

// ---------------------------------------------------------------------------
extern "C" void kernel_launch(void* const* d_in, const int* in_sizes, int n_in,
                              void* d_out, int out_size, void* d_ws, size_t ws_size,
                              hipStream_t stream) {
    const float* X    = (const float*)d_in[0];
    const float* A    = (const float*)d_in[1];
    const float* W    = (const float*)d_in[2];
    const float* bias = (const float*)d_in[3];
    float* out = (float*)d_out;

    const size_t HOPSZ = (size_t)BATCH * CIN * NNODES * SEQ;   // 33,554,432 elems

    float* ws    = (float*)d_ws;
    float* An    = ws;                       // 512*512 fp32        (1 MB)
    u16*   Afrag = (u16*)(ws + 262144);      // 32*16*512 bf16      (512 KB)
    u16*   Wfrag = Afrag + 262144;           // 4*4*512 bf16        (16 KB)
    u16*   Xb    = Wfrag + 8192;             // X in bf16           (64 MB)
    u16*   h1    = Xb + HOPSZ;
    u16*   h2    = h1 + HOPSZ;
    u16*   h3    = h2 + HOPSZ;

    mp_normalize<<<NNODES, 256, 0, stream>>>(A, An);
    mp_prep_afrag<<<dim3(16, 32), 32, 0, stream>>>(An, Afrag);
    mp_prep_wfrag<<<dim3(4, 4), 32, 0, stream>>>(W, Wfrag);
    mp_xcopy<<<HOPSZ / (256 * 8), 256, 0, stream>>>(X, Xb);

    dim3 hgrid(16, BATCH * CIN);
    mixprop_hop<<<hgrid, 256, 0, stream>>>(Afrag, Xb, Xb, h1);
    mixprop_hop<<<hgrid, 256, 0, stream>>>(Afrag, h1, Xb, h2);
    mixprop_hop<<<hgrid, 256, 0, stream>>>(Afrag, h2, Xb, h3);

    dim3 cgrid((NNODES * SEQ) / 128, BATCH);
    mixprop_conv<<<cgrid, 256, 0, stream>>>(Xb, h1, h2, h3, Wfrag, bias, out);
}
